// Transformer_59966333386977
// MI455X (gfx1250) — matrix operand order repaired and checked by reading, not compiled, not run on hardware
//
#include <hip/hip_runtime.h>
#include <hip/hip_bf16.h>

// ---------------------------------------------------------------------------
// MI455X (gfx1250) transformer block, compute-bound -> lives on the WMMA pipe.
// f32 tensors staged to LDS as bf16 (native fptrunc -> v_cvt_pk_bf16_f32),
// v_wmma_f32_16x16x32_bf16 with f32 accumulation.  GEMM: 128x128 workgroup
// tile, BK=32, double-buffered LDS, one barrier per K-step, b128 global loads.
// Attention: flash-style online softmax (never materializes the 537 MB score
// tensor); workspace = 5 x 33.5 MB.
// ---------------------------------------------------------------------------

typedef __attribute__((ext_vector_type(16))) __bf16 v16bf;
typedef __attribute__((ext_vector_type(8)))  __bf16 v8bf;
typedef __attribute__((ext_vector_type(2)))  __bf16 v2bf;
typedef __attribute__((ext_vector_type(8)))  float  v8f;

__device__ __forceinline__ unsigned pack2_bf16(float a, float b) {
  v2bf p;
  p[0] = (__bf16)a;                       // fptrunc, RNE -> v_cvt_pk_bf16_f32
  p[1] = (__bf16)b;
  return __builtin_bit_cast(unsigned, p);
}

__device__ __forceinline__ v16bf make_afrag(v8bf lo, v8bf hi8) {
  return __builtin_shufflevector(lo, hi8, 0, 1, 2, 3, 4, 5, 6, 7,
                                           8, 9, 10, 11, 12, 13, 14, 15);
}

// ---------------------------------------------------------------------------
// GEMM: C[M,N] = A[M,K] @ W[K,N] (+bias[N]) (+resid[M,N])
// 256 threads / 8 waves; WG tile 128x128, BK=32; wave tile 64x32 (8 WMMA/step)
// LDS double-buffered; A in [m][k], B transposed into [n][k] (K-contiguous).
// ---------------------------------------------------------------------------
__global__ __launch_bounds__(256)
void gemm_bf16_wmma(const float* __restrict__ A, const float* __restrict__ Bw,
                    const float* __restrict__ bias, const float* __restrict__ resid,
                    float* __restrict__ C, int M, int N, int K) {
  (void)M;
  __shared__ __align__(64) __bf16 Als[2][128][32];
  __shared__ __align__(64) __bf16 Bls[2][128][32];

  const int tid  = threadIdx.x;
  const int lane = tid & 31;
  const int wid  = tid >> 5;
  const int n0   = blockIdx.x * 128;
  const int m0   = blockIdx.y * 128;
  const int wm   = (wid & 1) * 64;     // 4 m-tiles
  const int wn   = (wid >> 1) * 32;    // 2 n-tiles

  // A staging: thread -> row ar (0..127), 16 contiguous k at base ac (0 or 16)
  const int ar = tid >> 1;
  const int ac = (tid & 1) * 16;
  // B staging: 2 units/thread, each = one k-row-pair x 4 consecutive n
  int bn4[2], bk2[2];
  const float* bP0[2];
  const float* bP1[2];
#pragma unroll
  for (int u = 0; u < 2; ++u) {
    int lin = u * 256 + tid;
    int kp  = lin >> 5;                 // 0..15 (k-row pair)
    int seg = lin & 31;                 // 0..31 (n segment of 4)
    bn4[u] = seg * 4;
    bk2[u] = kp * 2;
    bP0[u] = Bw + (size_t)(kp * 2) * N + n0 + bn4[u];
    bP1[u] = bP0[u] + N;
  }
  const float* aP = A + (size_t)(m0 + ar) * K + ac;

  const int row = lane & 15, hi = lane >> 4;
  const int kb = hi * 8, kh = hi * 16;

  v8f areg[2];
  float4 breg0[2], breg1[2];
  v8f acc[4][2] = {};

#define LOAD_REGS()                                                            \
  do {                                                                         \
    areg[0] = *(const v8f*)(aP);                                               \
    areg[1] = *(const v8f*)(aP + 8);                                           \
    _Pragma("unroll")                                                          \
    for (int u = 0; u < 2; ++u) {                                              \
      breg0[u] = *(const float4*)bP0[u];                                       \
      breg1[u] = *(const float4*)bP1[u];                                       \
      bP0[u] += (size_t)32 * N;                                                \
      bP1[u] += (size_t)32 * N;                                                \
    }                                                                          \
    aP += 32;                                                                  \
  } while (0)

#define STORE_LDS(BUF)                                                         \
  do {                                                                         \
    v8bf p0 = __builtin_convertvector(areg[0], v8bf);                          \
    v8bf p1 = __builtin_convertvector(areg[1], v8bf);                          \
    *(v8bf*)&Als[BUF][ar][ac]     = p0;                                        \
    *(v8bf*)&Als[BUF][ar][ac + 8] = p1;                                        \
    _Pragma("unroll")                                                          \
    for (int u = 0; u < 2; ++u) {                                              \
      const float* f0 = (const float*)&breg0[u];                               \
      const float* f1 = (const float*)&breg1[u];                               \
      _Pragma("unroll")                                                        \
      for (int j = 0; j < 4; ++j)                                              \
        *(unsigned*)&Bls[BUF][bn4[u] + j][bk2[u]] = pack2_bf16(f0[j], f1[j]);  \
    }                                                                          \
  } while (0)

#define COMPUTE(BUF)                                                           \
  do {                                                                         \
    v16bf bfr[2];                                                              \
    _Pragma("unroll")                                                          \
    for (int ni = 0; ni < 2; ++ni)                                             \
      bfr[ni] = *(const v16bf*)&Bls[BUF][wn + ni * 16 + row][kh];              \
    _Pragma("unroll")                                                          \
    for (int mi = 0; mi < 4; ++mi) {                                           \
      v8bf lo  = *(const v8bf*)&Als[BUF][wm + mi * 16 + row][kh];              \
      v8bf hi8 = *(const v8bf*)&Als[BUF][wm + mi * 16 + row][kh + 8];          \
      v16bf afr = make_afrag(lo, hi8);                                         \
      _Pragma("unroll")                                                        \
      for (int ni = 0; ni < 2; ++ni)                                           \
        acc[mi][ni] = __builtin_amdgcn_wmma_f32_16x16x32_bf16(                 \
            false, afr, false, bfr[ni], (short)0, acc[mi][ni], false, false);  \
    }                                                                          \
  } while (0)

  LOAD_REGS();
  STORE_LDS(0);
  __syncthreads();

  int buf = 0;
  for (int k0 = 32; k0 < K; k0 += 32) {
    LOAD_REGS();                                  // next slab -> registers
    __builtin_prefetch(aP, 0, 1);                 // global_prefetch_b8 (k0+32)
    COMPUTE(buf);                                 // WMMA from current buffer
    STORE_LDS(buf ^ 1);                           // regs -> alternate buffer
    __syncthreads();
    buf ^= 1;
  }
  COMPUTE(buf);

#undef LOAD_REGS
#undef STORE_LDS
#undef COMPUTE

  // Epilogue: C layout -> VGPR r holds M = r + 8*hi, N = lane&15
#pragma unroll
  for (int mi = 0; mi < 4; ++mi)
#pragma unroll
    for (int ni = 0; ni < 2; ++ni) {
      const int nl = wn + ni * 16 + row;
      float bb = bias ? bias[n0 + nl] : 0.f;
#pragma unroll
      for (int r = 0; r < 8; ++r) {
        int ml = wm + mi * 16 + r + 8 * hi;
        size_t off = (size_t)(m0 + ml) * N + (n0 + nl);
        float vv = acc[mi][ni][r] + bb;
        if (resid) vv += resid[off];
        C[off] = vv;
      }
    }
}

// ---------------------------------------------------------------------------
// RMSNorm: one 256-thread workgroup per row (D = 2048 -> 2 float4 / thread).
// ---------------------------------------------------------------------------
__global__ __launch_bounds__(256)
void rmsnorm_kernel(const float* __restrict__ x, const float* __restrict__ scale,
                    float* __restrict__ out, int D) {
  __shared__ float red[256];
  const int row = blockIdx.x, tid = threadIdx.x;
  const float* xr = x + (size_t)row * D;
  float s = 0.f;
  for (int j = tid * 4; j < D; j += 1024) {
    float4 v = *(const float4*)&xr[j];
    s += v.x * v.x + v.y * v.y + v.z * v.z + v.w * v.w;
  }
  red[tid] = s;
  __syncthreads();
  for (int off = 128; off > 0; off >>= 1) {
    if (tid < off) red[tid] += red[tid + off];
    __syncthreads();
  }
  float inv = rsqrtf(red[0] / (float)D + 1e-5f);
  float* orow = out + (size_t)row * D;
  for (int j = tid * 4; j < D; j += 1024) {
    float4 v = *(const float4*)&xr[j];
    float4 sc = *(const float4*)&scale[j];
    float4 o;
    o.x = sc.x * v.x * inv; o.y = sc.y * v.y * inv;
    o.z = sc.z * v.z * inv; o.w = sc.w * v.w * inv;
    *(float4*)&orow[j] = o;
  }
}

// ---------------------------------------------------------------------------
// Flash attention: WG = (b, h, 64-query tile); S = QK^T and O += P V on WMMA,
// online softmax in LDS (256-thread parallel, 4 segments/row).  No causal mask.
// ---------------------------------------------------------------------------
__global__ __launch_bounds__(256)
void flash_attn_wmma(const float* __restrict__ q, const float* __restrict__ k,
                     const float* __restrict__ v, float* __restrict__ o) {
  constexpr int T = 2048, D = 2048, HD = 128;
  __shared__ __align__(64) __bf16 Qs[64][128];    // [q][d]
  __shared__ __align__(64) __bf16 Ks[64][128];    // [kt][d]   B-frag for QK^T
  __shared__ __align__(64) __bf16 Vts[128][64];   // [d][kt]   B-frag for PV
  __shared__ __align__(64) float  Sf[64][64];
  __shared__ __align__(64) __bf16 Pbf[64][64];
  __shared__ __align__(32) float  m_i[64], l_i[64], alph[64];
  __shared__ float pred[64][4];

  const int tid = threadIdx.x, lane = tid & 31, wid = tid >> 5;
  const int qt = blockIdx.x, h = blockIdx.y, b = blockIdx.z;
  const size_t base = (size_t)b * T * D + (size_t)h * HD;
  const float qscale = 0.08838834764831845f;      // 1/sqrt(HD)

  const int row = lane & 15, hi = lane >> 4;
  const int kb = hi * 8, kh = hi * 16;
  const int qs = (wid >> 1) * 16;                 // PV q-strip (16 rows)
  const int ch = (wid & 1) * 64;                  // PV d-column half
  const int qi  = wid >> 1;                       // S tile q index (shared)
  const int ki0 = (wid & 1) * 2;                  // S tile k indices ki0, ki0+1

  // --- stage Q tile (pre-scaled by 1/sqrt(HD)); 32 floats/thread
  const int rk = tid >> 2;                        // row 0..63
  const int db = (tid & 3) * 32;                  // d base
  {
    const float* qP = q + base + (size_t)(qt * 64 + rk) * D + db;
#pragma unroll
    for (int sg = 0; sg < 4; ++sg) {
      v8f f = *(const v8f*)(qP + sg * 8);
      f *= qscale;
      *(v8bf*)&Qs[rk][db + sg * 8] = __builtin_convertvector(f, v8bf);
    }
  }
  if (tid < 64) { m_i[tid] = -3.0e38f; l_i[tid] = 0.f; }

  // K staging pointer (advance 64 rows per iteration)
  const float* kP = k + base + (size_t)rk * D + db;
  // V staging: 4 units/thread, each = one kt-row-pair x 4 consecutive d
  int vd4[4], vr2[4];
  const float* vP[4];
#pragma unroll
  for (int u = 0; u < 4; ++u) {
    int lin = u * 256 + tid;
    int rp  = lin >> 5;                 // 0..31 (kt row pair)
    int seg = lin & 31;                 // 0..31 (d segment of 4)
    vd4[u] = seg * 4;
    vr2[u] = rp * 2;
    vP[u]  = v + base + (size_t)(rp * 2) * D + vd4[u];
  }

  const int srow = tid >> 2;            // softmax row 0..63
  const int sc   = (tid & 3) * 16;      // softmax col segment

  v8f acc[4] = {};
  __syncthreads();

  for (int kt = 0; kt < T / 64; ++kt) {
    // ---- stage K (row-major) and V (transposed) tiles -------------------
#pragma unroll
    for (int sg = 0; sg < 4; ++sg) {
      v8f f = *(const v8f*)(kP + sg * 8);
      *(v8bf*)&Ks[rk][db + sg * 8] = __builtin_convertvector(f, v8bf);
    }
    kP += (size_t)64 * D;
#pragma unroll
    for (int u = 0; u < 4; ++u) {
      float4 f0 = *(const float4*)vP[u];
      float4 f1 = *(const float4*)(vP[u] + D);
      vP[u] += (size_t)64 * D;
      const float* a0 = (const float*)&f0;
      const float* a1 = (const float*)&f1;
#pragma unroll
      for (int j = 0; j < 4; ++j)
        *(unsigned*)&Vts[vd4[u] + j][vr2[u]] = pack2_bf16(a0[j], a1[j]);
    }
    __syncthreads();

    // ---- S = Q K^T : wave computes tiles (qi, ki0) and (qi, ki0+1) ------
    {
      v8f s0 = {}, s1 = {};
#pragma unroll
      for (int dk = 0; dk < 4; ++dk) {
        v8bf lo  = *(const v8bf*)&Qs[qi * 16 + row][dk * 32 + kh];
        v8bf hi8 = *(const v8bf*)&Qs[qi * 16 + row][dk * 32 + kh + 8];
        v16bf afr = make_afrag(lo, hi8);
        v16bf b0 = *(const v16bf*)&Ks[ki0 * 16 + row][dk * 32 + kh];
        v16bf b1 = *(const v16bf*)&Ks[(ki0 + 1) * 16 + row][dk * 32 + kh];
        s0 = __builtin_amdgcn_wmma_f32_16x16x32_bf16(false, afr, false, b0,
                                                     (short)0, s0, false, false);
        s1 = __builtin_amdgcn_wmma_f32_16x16x32_bf16(false, afr, false, b1,
                                                     (short)0, s1, false, false);
      }
#pragma unroll
      for (int r = 0; r < 8; ++r) {
        Sf[qi * 16 + r + 8 * hi][ki0 * 16 + row]       = s0[r];
        Sf[qi * 16 + r + 8 * hi][(ki0 + 1) * 16 + row] = s1[r];
      }
    }
    __syncthreads();

    // ---- online softmax: pass 1 (row max, 4 segments/row) ---------------
    {
      float mx = -3.0e38f;
#pragma unroll
      for (int j = 0; j < 16; ++j) mx = fmaxf(mx, Sf[srow][sc + j]);
      pred[srow][tid & 3] = mx;
    }
    __syncthreads();
    if (tid < 64) {
      float m_old = m_i[tid];
      float mnew = fmaxf(m_old,
                         fmaxf(fmaxf(pred[tid][0], pred[tid][1]),
                               fmaxf(pred[tid][2], pred[tid][3])));
      alph[tid] = __expf(m_old - mnew);
      m_i[tid]  = mnew;
    }
    __syncthreads();

    // ---- pass 2: P = exp(S - m), packed bf16 writes, partial row sums ---
    {
      float mnew = m_i[srow];
      float sum = 0.f;
#pragma unroll
      for (int j = 0; j < 16; j += 2) {
        float p0 = __expf(Sf[srow][sc + j]     - mnew);
        float p1 = __expf(Sf[srow][sc + j + 1] - mnew);
        sum += p0 + p1;
        *(unsigned*)&Pbf[srow][sc + j] = pack2_bf16(p0, p1);
      }
      pred[srow][tid & 3] = sum;
    }
    __syncthreads();
    if (tid < 64)
      l_i[tid] = alph[tid] * l_i[tid] +
                 (pred[tid][0] + pred[tid][1]) + (pred[tid][2] + pred[tid][3]);

    // ---- rescale running accumulators (one v8f LDS read), O += P V ------
    {
      v8f av = *(const v8f*)&alph[qs + 8 * hi];
#pragma unroll
      for (int nt = 0; nt < 4; ++nt)
#pragma unroll
        for (int r = 0; r < 8; ++r)
          acc[nt][r] *= av[r];
    }
#pragma unroll
    for (int ks = 0; ks < 2; ++ks) {
      v8bf lo  = *(const v8bf*)&Pbf[qs + row][ks * 32 + kh];
      v8bf hi8 = *(const v8bf*)&Pbf[qs + row][ks * 32 + kh + 8];
      v16bf afr = make_afrag(lo, hi8);
#pragma unroll
      for (int nt = 0; nt < 4; ++nt) {
        v16bf bf = *(const v16bf*)&Vts[ch + nt * 16 + row][ks * 32 + kh];
        acc[nt] = __builtin_amdgcn_wmma_f32_16x16x32_bf16(false, afr, false, bf,
                                                          (short)0, acc[nt], false, false);
      }
    }
    __syncthreads();
  }

  // ---- O = acc / l -> out[b, qrow, h*HD + d] ----------------------------
  {
    v8f lv = *(const v8f*)&l_i[qs + 8 * hi];
#pragma unroll
    for (int nt = 0; nt < 4; ++nt)
#pragma unroll
      for (int r = 0; r < 8; ++r) {
        int ml = qs + r + 8 * hi;
        o[base + (size_t)(qt * 64 + ml) * D + ch + nt * 16 + row] =
            acc[nt][r] * (1.f / lv[r]);
      }
  }
}

// ---------------------------------------------------------------------------
// SwiGLU-style gate: g = x1 * sigmoid(x1) * x2   (float4 vectorized)
// ---------------------------------------------------------------------------
__global__ void swiglu_kernel(const float* __restrict__ x1, const float* __restrict__ x2,
                              float* __restrict__ g, size_t n4) {
  size_t i = (size_t)blockIdx.x * blockDim.x + threadIdx.x;
  size_t stride = (size_t)gridDim.x * blockDim.x;
  for (; i < n4; i += stride) {
    float4 a = ((const float4*)x1)[i];
    float4 c = ((const float4*)x2)[i];
    float4 r;
    r.x = a.x * (1.f / (1.f + __expf(-a.x))) * c.x;
    r.y = a.y * (1.f / (1.f + __expf(-a.y))) * c.y;
    r.z = a.z * (1.f / (1.f + __expf(-a.z))) * c.z;
    r.w = a.w * (1.f / (1.f + __expf(-a.w))) * c.w;
    ((float4*)g)[i] = r;
  }
}

// ---------------------------------------------------------------------------
extern "C" void kernel_launch(void* const* d_in, const int* in_sizes, int n_in,
                              void* d_out, int out_size, void* d_ws, size_t ws_size,
                              hipStream_t stream) {
  (void)in_sizes; (void)n_in; (void)out_size; (void)ws_size;
  const float* x  = (const float*)d_in[0];
  const float* Wq = (const float*)d_in[1];
  const float* bq = (const float*)d_in[2];
  const float* Wk = (const float*)d_in[3];
  const float* bk = (const float*)d_in[4];
  const float* Wv = (const float*)d_in[5];
  const float* bv = (const float*)d_in[6];
  const float* Wo = (const float*)d_in[7];
  const float* bo = (const float*)d_in[8];
  const float* s1 = (const float*)d_in[9];
  const float* s2 = (const float*)d_in[10];
  const float* W1 = (const float*)d_in[11];
  const float* W2 = (const float*)d_in[12];
  const float* W3 = (const float*)d_in[13];
  float* out = (float*)d_out;

  constexpr int Bv = 2, Tv = 2048, Dv = 2048, NHv = 16;
  constexpr int Mv = Bv * Tv;
  const size_t NE = (size_t)Mv * Dv;            // 8.39M floats = 33.5 MB

  float* ws = (float*)d_ws;                     // 5 buffers, 167.8 MB total
  float* b0 = ws;                               // h1 -> h
  float* b1 = ws + NE;                          // q  -> h2 -> g
  float* b2 = ws + 2 * NE;                      // k  -> x1
  float* b3 = ws + 3 * NE;                      // v  -> x2
  float* b4 = ws + 4 * NE;                      // attn out

  dim3 gg(Dv / 128, Mv / 128);                  // 16 x 32 workgroups per GEMM
  dim3 ga(Tv / 64, NHv, Bv);                    // 32 x 16 x 2 attention WGs

  rmsnorm_kernel<<<Mv, 256, 0, stream>>>(x, s1, b0, Dv);                                // h1
  gemm_bf16_wmma<<<gg, 256, 0, stream>>>(b0, Wq, bq, nullptr, b1, Mv, Dv, Dv);          // q
  gemm_bf16_wmma<<<gg, 256, 0, stream>>>(b0, Wk, bk, nullptr, b2, Mv, Dv, Dv);          // k
  gemm_bf16_wmma<<<gg, 256, 0, stream>>>(b0, Wv, bv, nullptr, b3, Mv, Dv, Dv);          // v
  flash_attn_wmma<<<ga, 256, 0, stream>>>(b1, b2, b3, b4);                              // attn
  gemm_bf16_wmma<<<gg, 256, 0, stream>>>(b4, Wo, bo, x, b0, Mv, Dv, Dv);                // h = x + attn@Wo + bo
  rmsnorm_kernel<<<Mv, 256, 0, stream>>>(b0, s2, b1, Dv);                               // h2
  gemm_bf16_wmma<<<gg, 256, 0, stream>>>(b1, W1, nullptr, nullptr, b2, Mv, Dv, Dv);     // x1
  gemm_bf16_wmma<<<gg, 256, 0, stream>>>(b2, W2, nullptr, nullptr, b3, Mv, Dv, Dv);     // x2 = x1@W2
  swiglu_kernel<<<2048, 256, 0, stream>>>(b2, b3, b1, NE / 4);                          // g = x1*sig(x1)*x2
  gemm_bf16_wmma<<<gg, 256, 0, stream>>>(b1, W3, nullptr, x, out, Mv, Dv, Dv);          // out = g@W3 + x
}
